// MoELayer_45956150067562
// MI455X (gfx1250) — matrix operand order repaired and checked by reading, not compiled, not run on hardware
//
#include <hip/hip_runtime.h>
#include <hip/hip_bf16.h>
#include <math.h>

typedef __attribute__((ext_vector_type(16))) __bf16 v16bf;
typedef __attribute__((ext_vector_type(8)))  float  v8f;
typedef __attribute__((ext_vector_type(2)))  __bf16 bf16x2;
typedef __attribute__((ext_vector_type(4)))  __bf16 bf16x4;

#define NB 2
#define NS 2048
#define NH 1024
#define NE 8
#define NK 2
#define NF 4096
#define NT (NB*NS)      /* 4096 tokens */
#define NROWS (NT*NK)   /* 8192 (token,k) row slots */

// ---------------- CDNA5 async copy + WMMA primitives ----------------

// Per-lane async DMA: global memory -> LDS, 16 bytes/lane, tracked by ASYNCcnt.
__device__ __forceinline__ void async_b128(unsigned lds_byte_off, const void* gptr) {
  asm volatile("global_load_async_to_lds_b128 %0, %1, off"
               :: "v"(lds_byte_off), "v"(gptr)
               : "memory");
}
__device__ __forceinline__ void wait_async0() {
  asm volatile("s_wait_asynccnt 0x0" ::: "memory");
}
// Low 32 bits of a flat pointer into the LDS aperture == wave-relative LDS byte address.
__device__ __forceinline__ unsigned lds_off(const void* p) {
  return (unsigned)(uintptr_t)p;
}

__device__ __forceinline__ v8f wmma_bf16(v16bf a, v16bf b, v8f c) {
  return __builtin_amdgcn_wmma_f32_16x16x32_bf16(
      /*neg_a=*/false, a, /*neg_b=*/false, b,
      /*c_mod=*/(short)0, c, /*reuse_a=*/false, /*reuse_b=*/false);
}

// A fragment: 16x32 bf16 tile, row-major in LDS, stride 32.
// Lane L: m = L%16, half = L/16; VGPR v holds K pair (v>=4?16:0)+half*8+(v&3)*2.
__device__ __forceinline__ v16bf load_a_frag(const __bf16* base, int lane) {
  const int m = lane & 15, half = lane >> 4;
  const __bf16* row = base + m * 32;
  v16bf a;
#pragma unroll
  for (int v = 0; v < 8; ++v) {
    const int kb = ((v & 4) ? 16 : 0) + half * 8 + (v & 3) * 2;
    bf16x2 p = *(const bf16x2*)(row + kb);
    a[2 * v] = p.x;
    a[2 * v + 1] = p.y;
  }
  return a;
}

// B fragment: 32x16 bf16 tile stored transposed in LDS as Bs[n][k], stride 32.
// Lane L: n = L%16, half = L/16; VGPR v holds K pair (v>=4?16:0)+half*8+(v&3)*2.
__device__ __forceinline__ v16bf load_b_frag(const __bf16* base, int lane) {
  const int n = lane & 15, half = lane >> 4;
  const __bf16* row = base + n * 32 + half * 8;
  v16bf b;
#pragma unroll
  for (int v = 0; v < 8; ++v) {
    bf16x2 p = *(const bf16x2*)(row + ((v & 4) ? 16 : 0) + (v & 3) * 2);
    b[2 * v] = p.x;
    b[2 * v + 1] = p.y;
  }
  return b;
}

// ---------------- init / convert / transpose ----------------

__global__ void init_kernel(int* counts, int* counts2, float* zsum) {
  const int tid = threadIdx.x;
  if (tid < NE) { counts[tid] = 0; counts2[tid] = 0; }
  if (tid == 0) zsum[0] = 0.0f;
}

__global__ void cvt_x_kernel(const float* __restrict__ x, __bf16* __restrict__ y) {
  const int i = (blockIdx.x * blockDim.x + threadIdx.x) * 4;
  float4 v = *(const float4*)(x + i);
  bf16x4 o;
  o.x = (__bf16)v.x; o.y = (__bf16)v.y; o.z = (__bf16)v.z; o.w = (__bf16)v.w;
  *(bf16x4*)(y + i) = o;
}

// Per-expert transpose+convert: src f32 [R][C] -> dst bf16 [C][R].
__global__ __launch_bounds__(256)
void transpose_cvt_kernel(const float* __restrict__ src, __bf16* __restrict__ dst,
                          int R, int C) {
  __shared__ float tile[32][33];
  const int e = blockIdx.z;
  src += (size_t)e * R * C;
  dst += (size_t)e * R * C;
  const int c0 = blockIdx.x * 32, r0 = blockIdx.y * 32;
  const int tx = threadIdx.x & 31, ty = threadIdx.x >> 5;  // ty: 0..7
#pragma unroll
  for (int i = ty; i < 32; i += 8)
    tile[i][tx] = src[(size_t)(r0 + i) * C + (c0 + tx)];
  __syncthreads();
#pragma unroll
  for (int i = ty; i < 32; i += 8)
    dst[(size_t)(c0 + i) * R + (r0 + tx)] = (__bf16)tile[tx][i];
}

// ---------------- router: logits, softmax, top-2, losses ----------------

__global__ __launch_bounds__(256)
void router_kernel(const float* __restrict__ x, const float* __restrict__ Wr,
                   const float* __restrict__ br, int* __restrict__ topi,
                   float* __restrict__ wts, int* __restrict__ counts,
                   float* __restrict__ zsum) {
  const int lane = threadIdx.x & 31;
  const int wave = threadIdx.x >> 5;
  const int t = blockIdx.x * 8 + wave;
  const float* xr = x + (size_t)t * NH;

  float acc[NE];
#pragma unroll
  for (int e = 0; e < NE; ++e) acc[e] = 0.0f;
  for (int h = lane; h < NH; h += 32) {
    const float xv = xr[h];
#pragma unroll
    for (int e = 0; e < NE; ++e) acc[e] += xv * Wr[h * NE + e];
  }
#pragma unroll
  for (int off = 16; off > 0; off >>= 1) {
#pragma unroll
    for (int e = 0; e < NE; ++e) acc[e] += __shfl_xor(acc[e], off, 32);
  }
  if (lane == 0) {
    float mx = -1e30f;
#pragma unroll
    for (int e = 0; e < NE; ++e) { acc[e] += br[e]; mx = fmaxf(mx, acc[e]); }
    float p[NE], se = 0.0f;
#pragma unroll
    for (int e = 0; e < NE; ++e) { p[e] = expf(acc[e] - mx); se += p[e]; }
    int i1 = 0; float v1 = p[0];
#pragma unroll
    for (int e = 1; e < NE; ++e) if (p[e] > v1) { v1 = p[e]; i1 = e; }
    int i2 = -1; float v2 = -1.0f;
#pragma unroll
    for (int e = 0; e < NE; ++e) if (e != i1 && p[e] > v2) { v2 = p[e]; i2 = e; }
    const float inv = 1.0f / (v1 + v2);
    topi[2 * t] = i1;     topi[2 * t + 1] = i2;
    wts[2 * t] = v1 * inv; wts[2 * t + 1] = v2 * inv;
    atomicAdd(&counts[i1], 1);
    atomicAdd(&counts[i2], 1);
    atomicAdd(zsum, logf(se) + mx);   // logsumexp of logits
  }
}

// ---------------- gather per-expert row lists ----------------

__global__ void gather_kernel(const int* __restrict__ topi, int* __restrict__ counts2,
                              int* __restrict__ rowlist) {
  const int t = blockIdx.x * blockDim.x + threadIdx.x;
#pragma unroll
  for (int k = 0; k < NK; ++k) {
    const int e = topi[2 * t + k];
    const int pos = atomicAdd(&counts2[e], 1);
    rowlist[e * NT + pos] = 2 * t + k;
  }
}

// ---------------- grouped GEMM, async-pipelined double buffer ----------------
// Block tile 128(M) x 64(N), K-step 32. 8 waves as 4(M) x 2(N), wave tile 32x32.
// Stage: A tile 128x32 bf16 (8KB) + B tile 64rows x 32 bf16 (4KB), both via
// global_load_async_to_lds_b128 (3 async instructions per wave per stage).

#define GEMM_PIPELINE(A_SRC_ROW_PTR, KDIM, BT_PTR)                                   \
  unsigned ldsA = lds_off(&As[0][0]);                                                \
  unsigned ldsB = lds_off(&Bs[0][0]);                                                \
  const int arow = tid >> 1, ahf = tid & 1;                                          \
  const int aslot = rids[arow];                                                      \
  const __bf16* aptr = (A_SRC_ROW_PTR) + ahf * 16;                                   \
  const unsigned adst = (unsigned)(arow * 32 + ahf * 16) * 2;                        \
  const int brow = tid >> 2, bseg = tid & 3;                                         \
  const __bf16* bptr = (BT_PTR) + (size_t)(n0 + brow) * (KDIM) + bseg * 8;           \
  const unsigned bdst = (unsigned)(brow * 32 + bseg * 8) * 2;                        \
  /* prologue: stage 0 */                                                            \
  async_b128(ldsA + adst,      aptr);                                                \
  async_b128(ldsA + adst + 16, aptr + 8);                                            \
  async_b128(ldsB + bdst,      bptr);                                                \
  const int niter = (KDIM) / 32;                                                     \
  for (int it = 0; it < niter; ++it) {                                               \
    wait_async0();                                                                   \
    __syncthreads();                                                                 \
    if (it + 1 < niter) {                                                            \
      const int k1 = (it + 1) * 32;                                                  \
      const unsigned bo = (unsigned)((it + 1) & 1);                                  \
      async_b128(ldsA + bo * 8192 + adst,      aptr + k1);                           \
      async_b128(ldsA + bo * 8192 + adst + 16, aptr + k1 + 8);                       \
      async_b128(ldsB + bo * 4096 + bdst,      bptr + k1);                           \
    }                                                                                \
    const __bf16* Ab = &As[it & 1][0];                                               \
    const __bf16* Bb = &Bs[it & 1][0];                                               \
    v16bf a0 = load_a_frag(Ab + (wm * 32) * 32, lane);                               \
    v16bf a1 = load_a_frag(Ab + (wm * 32 + 16) * 32, lane);                          \
    v16bf b0 = load_b_frag(Bb + (wn * 32) * 32, lane);                               \
    v16bf b1v = load_b_frag(Bb + (wn * 32 + 16) * 32, lane);                         \
    acc00 = wmma_bf16(a0, b0, acc00);                                                \
    acc01 = wmma_bf16(a0, b1v, acc01);                                               \
    acc10 = wmma_bf16(a1, b0, acc10);                                                \
    acc11 = wmma_bf16(a1, b1v, acc11);                                               \
  }

// FFN1: H1 = gelu(Xg @ W1t[e]^T + b1[e]);  A rows from gathered X, B from W1t [F][H].
__global__ __launch_bounds__(256)
void ffn1_kernel(const __bf16* __restrict__ Xbf, const __bf16* __restrict__ W1t,
                 const float* __restrict__ b1, const int* __restrict__ rowlist,
                 const int* __restrict__ cnts, __bf16* __restrict__ H1) {
  __shared__ __bf16 As[2][128 * 32];
  __shared__ __bf16 Bs[2][64 * 32];
  __shared__ int rids[128];

  const int e = blockIdx.z;
  const int cnt = cnts[e];
  const int m0 = blockIdx.x * 128;
  if (m0 >= cnt) return;
  const int n0 = blockIdx.y * 64;
  const int tid = threadIdx.x;
  const int lane = tid & 31;
  const int wid = tid >> 5;
  const int wm = wid >> 1, wn = wid & 1;

  if (tid < 128) {
    const int g = m0 + tid;
    rids[tid] = (g < cnt) ? rowlist[e * NT + g] : -1;
  }
  __syncthreads();

  v8f acc00 = {0,0,0,0,0,0,0,0}, acc01 = {0,0,0,0,0,0,0,0};
  v8f acc10 = {0,0,0,0,0,0,0,0}, acc11 = {0,0,0,0,0,0,0,0};

  GEMM_PIPELINE(Xbf + (size_t)((aslot < 0) ? 0 : (aslot >> 1)) * NH, NH,
                W1t + (size_t)e * NF * NH)

  // Epilogue: bias + exact GELU -> bf16 H1
  const int nl = lane & 15, hf = lane >> 4;
#pragma unroll
  for (int tm = 0; tm < 2; ++tm) {
#pragma unroll
    for (int tn = 0; tn < 2; ++tn) {
      v8f c = (tm == 0) ? ((tn == 0) ? acc00 : acc01) : ((tn == 0) ? acc10 : acc11);
      const int f = n0 + wn * 32 + tn * 16 + nl;
      const float bb = b1[(size_t)e * NF + f];
#pragma unroll
      for (int r = 0; r < 8; ++r) {
        const int ml = wm * 32 + tm * 16 + hf * 8 + r;
        const int rid = rids[ml];
        if (rid >= 0) {
          const float v = c[r] + bb;
          const float g = 0.5f * v * (1.0f + erff(v * 0.70710678118f));
          H1[(size_t)rid * NF + f] = (__bf16)g;
        }
      }
    }
  }
}

// FFN2: Y = H1 @ W2t[e]^T + b2[e];  A rows from H1, B from W2t [H][F].
__global__ __launch_bounds__(256)
void ffn2_kernel(const __bf16* __restrict__ H1, const __bf16* __restrict__ W2t,
                 const float* __restrict__ b2, const int* __restrict__ rowlist,
                 const int* __restrict__ cnts, float* __restrict__ Ybuf) {
  __shared__ __bf16 As[2][128 * 32];
  __shared__ __bf16 Bs[2][64 * 32];
  __shared__ int rids[128];

  const int e = blockIdx.z;
  const int cnt = cnts[e];
  const int m0 = blockIdx.x * 128;
  if (m0 >= cnt) return;
  const int n0 = blockIdx.y * 64;
  const int tid = threadIdx.x;
  const int lane = tid & 31;
  const int wid = tid >> 5;
  const int wm = wid >> 1, wn = wid & 1;

  if (tid < 128) {
    const int g = m0 + tid;
    rids[tid] = (g < cnt) ? rowlist[e * NT + g] : -1;
  }
  __syncthreads();

  v8f acc00 = {0,0,0,0,0,0,0,0}, acc01 = {0,0,0,0,0,0,0,0};
  v8f acc10 = {0,0,0,0,0,0,0,0}, acc11 = {0,0,0,0,0,0,0,0};

  GEMM_PIPELINE(H1 + (size_t)((aslot < 0) ? 0 : aslot) * NF, NF,
                W2t + (size_t)e * NH * NF)

  const int nl = lane & 15, hf = lane >> 4;
#pragma unroll
  for (int tm = 0; tm < 2; ++tm) {
#pragma unroll
    for (int tn = 0; tn < 2; ++tn) {
      v8f c = (tm == 0) ? ((tn == 0) ? acc00 : acc01) : ((tn == 0) ? acc10 : acc11);
      const int h = n0 + wn * 32 + tn * 16 + nl;
      const float bb = b2[(size_t)e * NH + h];
#pragma unroll
      for (int r = 0; r < 8; ++r) {
        const int ml = wm * 32 + tm * 16 + hf * 8 + r;
        const int rid = rids[ml];
        if (rid >= 0) Ybuf[(size_t)rid * NH + h] = c[r] + bb;
      }
    }
  }
}

// ---------------- combine: out[t] = w0*Y[2t] + w1*Y[2t+1] ----------------

__global__ void combine_kernel(const float* __restrict__ Ybuf, const float* __restrict__ wts,
                               float* __restrict__ out) {
  const int i = blockIdx.x * blockDim.x + threadIdx.x;
  const int t = i >> 10;        // / NH
  const int h = i & (NH - 1);
  out[i] = wts[2 * t] * Ybuf[(size_t)(2 * t) * NH + h] +
           wts[2 * t + 1] * Ybuf[(size_t)(2 * t + 1) * NH + h];
}

// ---------------- aux + z loss scalar ----------------

__global__ void loss_kernel(const int* __restrict__ counts, const float* __restrict__ zsum,
                            float* __restrict__ out_loss) {
  const int tid = threadIdx.x;
  float p = 0.0f;
  if (tid < NE) {
    const float load = (float)counts[tid] / (float)(NT * NK);
    const float d = load - (1.0f / NE);
    p = d * d;
  }
#pragma unroll
  for (int off = 16; off > 0; off >>= 1) p += __shfl_xor(p, off, 32);
  if (tid == 0)
    out_loss[0] = 0.01f * (p / NE) + 0.001f * (zsum[0] / (float)NT);
}

// ---------------- launch ----------------

extern "C" void kernel_launch(void* const* d_in, const int* in_sizes, int n_in,
                              void* d_out, int out_size, void* d_ws, size_t ws_size,
                              hipStream_t stream) {
  (void)in_sizes; (void)n_in; (void)out_size; (void)ws_size;
  const float* x  = (const float*)d_in[0];
  const float* Wr = (const float*)d_in[1];
  const float* br = (const float*)d_in[2];
  const float* W1 = (const float*)d_in[3];
  const float* b1 = (const float*)d_in[4];
  const float* W2 = (const float*)d_in[5];
  const float* b2 = (const float*)d_in[6];
  float* out = (float*)d_out;
  float* out_loss = out + (size_t)NT * NH;

  char* w = (char*)d_ws;
  size_t off = 0;
  auto carve = [&](size_t bytes) -> void* {
    void* p = w + off;
    off += (bytes + 255) & ~(size_t)255;
    return p;
  };
  __bf16* Xbf    = (__bf16*)carve((size_t)NT * NH * 2);        //   8 MB
  __bf16* W1t    = (__bf16*)carve((size_t)NE * NH * NF * 2);   //  64 MB  [E][F][H]
  __bf16* W2t    = (__bf16*)carve((size_t)NE * NF * NH * 2);   //  64 MB  [E][H][F]
  __bf16* H1     = (__bf16*)carve((size_t)NROWS * NF * 2);     //  64 MB
  float*  Ybuf   = (float*) carve((size_t)NROWS * NH * 4);     //  32 MB
  int*    topi   = (int*)   carve((size_t)NT * NK * 4);
  float*  wts    = (float*) carve((size_t)NT * NK * 4);
  int*    counts  = (int*)  carve(NE * 4);
  int*    counts2 = (int*)  carve(NE * 4);
  float*  zsum   = (float*) carve(4);
  int*    rowlist = (int*)  carve((size_t)NE * NT * 4);

  init_kernel<<<1, 64, 0, stream>>>(counts, counts2, zsum);
  cvt_x_kernel<<<(NT * NH / 4) / 256, 256, 0, stream>>>(x, Xbf);
  transpose_cvt_kernel<<<dim3(NF / 32, NH / 32, NE), 256, 0, stream>>>(W1, W1t, NH, NF);
  transpose_cvt_kernel<<<dim3(NH / 32, NF / 32, NE), 256, 0, stream>>>(W2, W2t, NF, NH);
  router_kernel<<<NT / 8, 256, 0, stream>>>(x, Wr, br, topi, wts, counts, zsum);
  gather_kernel<<<NT / 256, 256, 0, stream>>>(topi, counts2, rowlist);
  ffn1_kernel<<<dim3(NT / 128, NF / 64, NE), 256, 0, stream>>>(Xbf, W1t, b1, rowlist, counts2, H1);
  ffn2_kernel<<<dim3(NT / 128, NH / 64, NE), 256, 0, stream>>>(H1, W2t, b2, rowlist, counts2, Ybuf);
  combine_kernel<<<(NT * NH) / 256, 256, 0, stream>>>(Ybuf, wts, out);
  loss_kernel<<<1, 32, 0, stream>>>(counts, zsum, out_loss);
}
